// InjectAttentionProcessor_54262616818231
// MI455X (gfx1250) — matrix operand order repaired and checked by reading, not compiled, not run on hardware
//
#include <hip/hip_runtime.h>
#include <hip/hip_bf16.h>

// ---------------------------------------------------------------------------
// Fused multi-head self-attention for MI455X (gfx1250, wave32, WMMA).
// B=2, S=4096, C=640, H=8, D=80.  All matmuls via v_wmma_f32_16x16x32_bf16.
// Pipeline: cvt(x) + cvt/transpose(W) -> QKV proj -> flash attention -> out proj.
// All WMMA operands are pre-converted bf16 with layouts chosen so every
// fragment is two 16-byte vector loads (no per-element conversion in hot loops).
// Attention uses 32 query rows per wave to halve K/V L2 traffic per FLOP.
// ---------------------------------------------------------------------------

typedef __bf16 bf16_t;
typedef bf16_t v16bf __attribute__((ext_vector_type(16)));
typedef float  v8f   __attribute__((ext_vector_type(8)));

#define BATCH 2
#define SEQ   4096
#define CH    640
#define NH    8
#define HD    80
#define HDP   96          // head_dim padded to 3*32 (zero-filled) for Q/K

// ---- scalar helpers -------------------------------------------------------

static __device__ __forceinline__ unsigned short f2bfu(float f) {
  union { float f; unsigned int u; } x; x.f = f;
  unsigned int r = x.u + 0x7FFFu + ((x.u >> 16) & 1u);   // RNE
  return (unsigned short)(r >> 16);
}
static __device__ __forceinline__ bf16_t f2bf(float f) {
  unsigned short h = f2bfu(f);
  return __builtin_bit_cast(bf16_t, h);
}
static __device__ __forceinline__ bf16_t bfzero() {
  return __builtin_bit_cast(bf16_t, (unsigned short)0);
}
static __device__ __forceinline__ unsigned int pack2(float lo, float hi) {
  return (unsigned int)f2bfu(lo) | ((unsigned int)f2bfu(hi) << 16);
}

// ---- WMMA fragments -------------------------------------------------------

union frag16 { v16bf v; uint4 q[2]; };

// A-fragment (16x32, ISA A layout): elem(m = lane&15, k) = src[m*ld + k]
//   slots 0..3: k = 8*(l>>4) + 0..7 ; slots 4..7: k = 16 + 8*(l>>4) + 0..7
// Requires src 16B-aligned and ld a multiple of 8 elements.
static __device__ __forceinline__ v16bf load_frag_a(const bf16_t* src, int ld) {
  const int lane = threadIdx.x & 31;
  const bf16_t* row = src + (size_t)(lane & 15) * ld + ((lane >> 4) << 3);
  frag16 f;
  f.q[0] = *(const uint4*)(row);        // k = ksel + 0..7
  f.q[1] = *(const uint4*)(row + 16);   // k = 16 + ksel + 0..7
  return f.v;
}

// B-fragment (32x16, ISA B layout) loaded from a TRANSPOSED row-major source:
// elem(k, n = lane&15) = src[n*ld + k];  same k order as load_frag_a.
static __device__ __forceinline__ v16bf load_frag_bT(const bf16_t* src, int ld) {
  const int lane = threadIdx.x & 31;
  const bf16_t* row = src + (size_t)(lane & 15) * ld + ((lane >> 4) << 3);
  frag16 f;
  f.q[0] = *(const uint4*)(row);        // k = ksel + 0..7
  f.q[1] = *(const uint4*)(row + 16);   // k = 16 + ksel + 0..7
  return f.v;
}

static __device__ __forceinline__ v8f wmma_bf16(v16bf a, v16bf b, v8f c) {
  return __builtin_amdgcn_wmma_f32_16x16x32_bf16(false, a, false, b, (short)0, c, false, false);
}

// reductions across the 16 lanes holding one C-matrix row (xor stays in-group)
static __device__ __forceinline__ float rowmax16(float x) {
  x = fmaxf(x, __shfl_xor(x, 1, 32));
  x = fmaxf(x, __shfl_xor(x, 2, 32));
  x = fmaxf(x, __shfl_xor(x, 4, 32));
  x = fmaxf(x, __shfl_xor(x, 8, 32));
  return x;
}
static __device__ __forceinline__ float rowsum16(float x) {
  x += __shfl_xor(x, 1, 32);
  x += __shfl_xor(x, 2, 32);
  x += __shfl_xor(x, 4, 32);
  x += __shfl_xor(x, 8, 32);
  return x;
}

// ---------------------------------------------------------------------------
// Stage 0a: x fp32 -> bf16 (one pass; removes all conversions from GEMM loops)
// ---------------------------------------------------------------------------
__global__ __launch_bounds__(256) void cvt_x_kernel(const float* __restrict__ in,
                                                    bf16_t* __restrict__ out) {
  const size_t idx = ((size_t)blockIdx.x * 256 + threadIdx.x) * 8;
  const float4* p = (const float4*)(in + idx);
  float4 f0 = p[0], f1 = p[1];
  uint4 u;
  u.x = pack2(f0.x, f0.y); u.y = pack2(f0.z, f0.w);
  u.z = pack2(f1.x, f1.y); u.w = pack2(f1.z, f1.w);
  *(uint4*)(out + idx) = u;
}

// ---------------------------------------------------------------------------
// Stage 0b: W fp32 [c][d] -> bf16 transposed Wt [d][c] (LDS-tiled transpose)
// ---------------------------------------------------------------------------
__global__ __launch_bounds__(256) void cvt_w_kernel(const float* __restrict__ W,
                                                    bf16_t* __restrict__ Wt) {
  __shared__ float tile[16][17];
  const int tx = threadIdx.x, ty = threadIdx.y;
  const int d0 = blockIdx.x * 16, c0 = blockIdx.y * 16;
  tile[ty][tx] = W[(size_t)(c0 + ty) * CH + d0 + tx];
  __syncthreads();
  Wt[(size_t)(d0 + ty) * CH + c0 + tx] = f2bf(tile[tx][ty]);
}

// ---------------------------------------------------------------------------
// Stage 1: Q/K/V projection.  One wave computes a 64x32 output block
// (4 M-tiles x 2 N-tiles = 8 wmma per K-step).  grid = (8192/64, 20/4).
//   mode 0: out = head-split row-major [bh][s][96], pad cols 80..95 zeroed
//   mode 1: out = head-split TRANSPOSED [bh][d(80)][s]  (for V; 16B packed stores)
// ---------------------------------------------------------------------------
__global__ __launch_bounds__(128) void proj_kernel(
    const bf16_t* __restrict__ xbf, const bf16_t* __restrict__ Wt,
    bf16_t* __restrict__ out, float scale, int v_transposed) {
  const int lane = threadIdx.x & 31;
  const int wid  = threadIdx.x >> 5;
  const int row0 = blockIdx.x * 64;                 // b*S+s base
  const int col0 = (blockIdx.y * 4 + wid) * 32;     // output channel base

  v8f acc[4][2] = {};

  for (int kk = 0; kk < CH; kk += 32) {
    v16bf a[4], b[2];
#pragma unroll
    for (int mt = 0; mt < 4; ++mt)
      a[mt] = load_frag_a(xbf + (size_t)(row0 + mt * 16) * CH + kk, CH);
#pragma unroll
    for (int nt = 0; nt < 2; ++nt)
      b[nt] = load_frag_bT(Wt + (size_t)(col0 + nt * 16) * CH + kk, CH);
#pragma unroll
    for (int mt = 0; mt < 4; ++mt)
#pragma unroll
      for (int nt = 0; nt < 2; ++nt)
        acc[mt][nt] = wmma_bf16(a[mt], b[nt], acc[mt][nt]);
  }

  const int n  = lane & 15;
  const int mh = (lane >> 4) * 8;
#pragma unroll
  for (int mt = 0; mt < 4; ++mt) {
#pragma unroll
    for (int nt = 0; nt < 2; ++nt) {
      const int cb = col0 + nt * 16;
      const int h = cb / HD, d0 = cb % HD;   // 16-wide tile never straddles a head
      if (!v_transposed) {
#pragma unroll
        for (int r = 0; r < 8; ++r) {
          int gr = row0 + mt * 16 + mh + r;
          int b_ = gr >> 12, s = gr & (SEQ - 1);
          bf16_t* o = out + (((size_t)(b_ * NH + h) * SEQ) + s) * HDP + d0;
          o[n] = f2bf(acc[mt][nt][r] * scale);
          if (d0 == 64) o[16 + n] = bfzero();   // zero the 80..95 pad
        }
      } else {
        // V: store transposed [bh][d][s]; 8 consecutive s per lane -> one b128
        int sb = row0 + mt * 16 + mh;
        int b_ = sb >> 12, s = sb & (SEQ - 1);
        const v8f& c = acc[mt][nt];
        uint4 u;
        u.x = pack2(c[0], c[1]); u.y = pack2(c[2], c[3]);
        u.z = pack2(c[4], c[5]); u.w = pack2(c[6], c[7]);
        *(uint4*)(out + ((size_t)(b_ * NH + h) * HD + d0 + n) * SEQ + s) = u;
      }
    }
  }
}

// ---------------------------------------------------------------------------
// Stage 2: flash attention.  One wave owns a 32-row query tile (2 M-tiles),
// streams keys 32 at a time; scores stay in registers (online softmax).
// Each K/V fragment load feeds 2 wmma -> wmma:b128 ratio 1.0.
// grid = (SEQ/128, 16).  Q/K: [bh][s][96] zero-padded, V: [bh][d][s].
// ---------------------------------------------------------------------------
__global__ __launch_bounds__(128) void attn_kernel(
    const bf16_t* __restrict__ qb, const bf16_t* __restrict__ kb,
    const bf16_t* __restrict__ vt, bf16_t* __restrict__ attn) {
  __shared__ bf16_t plds[4 * 2 * 16 * 32];    // per-wave 2x(16x32) P staging, 8 KB
  const int lane = threadIdx.x & 31;
  const int wid  = threadIdx.x >> 5;
  const int bh   = blockIdx.y;
  const int q0   = (blockIdx.x * 4 + wid) * 32;
  const size_t baseQK = (size_t)bh * SEQ * HDP;
  const size_t baseV  = (size_t)bh * HD * SEQ;

  // resident Q fragments: 2 q-tiles x 3 k-steps (scale pre-folded, pad zeroed)
  v16bf aq[2][3];
#pragma unroll
  for (int qi = 0; qi < 2; ++qi) {
    const bf16_t* qrow = qb + baseQK + (size_t)(q0 + qi * 16) * HDP;
#pragma unroll
    for (int i = 0; i < 3; ++i) aq[qi][i] = load_frag_a(qrow + 32 * i, HDP);
  }

  v8f o[2][5] = {};
  float mrow[2][8], lrow[2][8];
#pragma unroll
  for (int qi = 0; qi < 2; ++qi)
#pragma unroll
    for (int r = 0; r < 8; ++r) { mrow[qi][r] = -1e30f; lrow[qi][r] = 0.f; }

  bf16_t* pl = plds + wid * (2 * 16 * 32);

#pragma unroll 1
  for (int j0 = 0; j0 < SEQ; j0 += 32) {
    // S = Q * K^T : each K fragment is consumed by both q-tiles immediately
    v8f s[2][2] = {};
#pragma unroll
    for (int kt = 0; kt < 2; ++kt) {
      const bf16_t* kr = kb + baseQK + (size_t)(j0 + kt * 16) * HDP;
#pragma unroll
      for (int i = 0; i < 3; ++i) {
        v16bf bk = load_frag_bT(kr + 32 * i, HDP);
        s[0][kt] = wmma_bf16(aq[0][i], bk, s[0][kt]);
        s[1][kt] = wmma_bf16(aq[1][i], bk, s[1][kt]);
      }
    }

    // online softmax + P staging (C-layout -> A-layout through per-wave LDS)
    const int n  = lane & 15;
    const int mh = (lane >> 4) * 8;
#pragma unroll
    for (int qi = 0; qi < 2; ++qi) {
#pragma unroll
      for (int r = 0; r < 8; ++r) {
        float mx   = rowmax16(fmaxf(s[qi][0][r], s[qi][1][r]));
        float mnew = fmaxf(mrow[qi][r], mx);
        float p0 = __expf(s[qi][0][r] - mnew);
        float p1 = __expf(s[qi][1][r] - mnew);
        float alpha = __expf(mrow[qi][r] - mnew);
        float rs = rowsum16(p0 + p1);
        lrow[qi][r] = lrow[qi][r] * alpha + rs;
        mrow[qi][r] = mnew;
        bf16_t* pq = pl + qi * (16 * 32);
        pq[(mh + r) * 32 + n]      = f2bf(p0);
        pq[(mh + r) * 32 + n + 16] = f2bf(p1);
#pragma unroll
        for (int t = 0; t < 5; ++t) o[qi][t][r] *= alpha;
      }
    }
    asm volatile("s_wait_dscnt 0" ::: "memory");
    v16bf ap0 = load_frag_a(pl, 32);
    v16bf ap1 = load_frag_a(pl + 16 * 32, 32);

    // O += P * V : each V fragment feeds both q-tiles
    const bf16_t* vr = vt + baseV + j0;
#pragma unroll
    for (int t = 0; t < 5; ++t) {
      v16bf bv = load_frag_bT(vr + (size_t)(t * 16) * SEQ, SEQ);
      o[0][t] = wmma_bf16(ap0, bv, o[0][t]);
      o[1][t] = wmma_bf16(ap1, bv, o[1][t]);
    }
  }

  // finalize: divide by denom, merge heads into bf16 [B*S][C]
  const int b_ = bh >> 3;
  const int h  = bh & (NH - 1);
  const int n  = lane & 15;
  const int mh = (lane >> 4) * 8;
#pragma unroll
  for (int qi = 0; qi < 2; ++qi) {
#pragma unroll
    for (int r = 0; r < 8; ++r) {
      float inv = 1.0f / lrow[qi][r];
      size_t grow = (size_t)b_ * SEQ + q0 + qi * 16 + mh + r;
      bf16_t* arow = attn + grow * CH + h * HD;
#pragma unroll
      for (int t = 0; t < 5; ++t) arow[t * 16 + n] = f2bf(o[qi][t][r] * inv);
    }
  }
}

// ---------------------------------------------------------------------------
// Stage 3: output projection + bias (fp32 out).  Same 64x32 blocking.
// ---------------------------------------------------------------------------
__global__ __launch_bounds__(128) void out_proj_kernel(
    const bf16_t* __restrict__ a_in, const bf16_t* __restrict__ Wot,
    const float* __restrict__ bo, float* __restrict__ out) {
  const int lane = threadIdx.x & 31;
  const int wid  = threadIdx.x >> 5;
  const int row0 = blockIdx.x * 64;
  const int col0 = (blockIdx.y * 4 + wid) * 32;
  const int n    = lane & 15;

  const float bias[2] = { bo[col0 + n], bo[col0 + 16 + n] };

  v8f acc[4][2] = {};
  for (int kk = 0; kk < CH; kk += 32) {
    v16bf a[4], b[2];
#pragma unroll
    for (int mt = 0; mt < 4; ++mt)
      a[mt] = load_frag_a(a_in + (size_t)(row0 + mt * 16) * CH + kk, CH);
#pragma unroll
    for (int nt = 0; nt < 2; ++nt)
      b[nt] = load_frag_bT(Wot + (size_t)(col0 + nt * 16) * CH + kk, CH);
#pragma unroll
    for (int mt = 0; mt < 4; ++mt)
#pragma unroll
      for (int nt = 0; nt < 2; ++nt)
        acc[mt][nt] = wmma_bf16(a[mt], b[nt], acc[mt][nt]);
  }

  const int mh = (lane >> 4) * 8;
#pragma unroll
  for (int mt = 0; mt < 4; ++mt)
#pragma unroll
    for (int nt = 0; nt < 2; ++nt)
#pragma unroll
      for (int r = 0; r < 8; ++r) {
        size_t gr = (size_t)row0 + mt * 16 + mh + r;
        out[gr * CH + col0 + nt * 16 + n] = acc[mt][nt][r] + bias[nt];
      }
}

// ---------------------------------------------------------------------------

extern "C" void kernel_launch(void* const* d_in, const int* in_sizes, int n_in,
                              void* d_out, int out_size, void* d_ws, size_t ws_size,
                              hipStream_t stream) {
  const float* x  = (const float*)d_in[0];
  const float* Wq = (const float*)d_in[1];
  const float* Wk = (const float*)d_in[2];
  const float* Wv = (const float*)d_in[3];
  const float* Wo = (const float*)d_in[4];
  const float* bo = (const float*)d_in[5];
  float* out = (float*)d_out;

  const size_t NX   = (size_t)BATCH * SEQ * CH;       // 5,242,880
  const size_t NW   = (size_t)CH * CH;                // 409,600
  const size_t NQK  = (size_t)BATCH * NH * SEQ * HDP; // 6,291,456
  const size_t NV   = (size_t)BATCH * NH * HD * SEQ;  // 5,242,880

  bf16_t* xbf = (bf16_t*)d_ws;
  bf16_t* Wqt = xbf + NX;
  bf16_t* Wkt = Wqt + NW;
  bf16_t* Wvt = Wkt + NW;
  bf16_t* Wot = Wvt + NW;
  bf16_t* qb  = Wot + NW;
  bf16_t* kbm = qb + NQK;
  bf16_t* vtb = kbm + NQK;
  bf16_t* at  = vtb + NV;

  dim3 b256(256, 1, 1);
  dim3 bw(16, 16, 1);
  dim3 gw(CH / 16, CH / 16, 1);                       // (40, 40)
  dim3 b128(128, 1, 1);
  dim3 gproj((BATCH * SEQ) / 64, (CH / 32) / 4, 1);   // (128, 5)
  dim3 gattn(SEQ / 128, BATCH * NH, 1);               // (32, 16)

  const float scale = 0.11180339887498949f;           // 1/sqrt(80), folded into Q

  cvt_x_kernel<<<dim3((unsigned)(NX / (256 * 8))), b256, 0, stream>>>(x, xbf);
  cvt_w_kernel<<<gw, bw, 0, stream>>>(Wq, Wqt);
  cvt_w_kernel<<<gw, bw, 0, stream>>>(Wk, Wkt);
  cvt_w_kernel<<<gw, bw, 0, stream>>>(Wv, Wvt);
  cvt_w_kernel<<<gw, bw, 0, stream>>>(Wo, Wot);

  proj_kernel<<<gproj, b128, 0, stream>>>(xbf, Wqt, qb, scale, 0);
  proj_kernel<<<gproj, b128, 0, stream>>>(xbf, Wkt, kbm, 1.0f, 0);
  proj_kernel<<<gproj, b128, 0, stream>>>(xbf, Wvt, vtb, 1.0f, 1);

  attn_kernel<<<gattn, b128, 0, stream>>>(qb, kbm, vtb, at);
  out_proj_kernel<<<gproj, b128, 0, stream>>>(at, Wot, bo, out);
}
